// GNN_60318520705487
// MI455X (gfx1250) — matrix operand order repaired and checked by reading, not compiled, not run on hardware
//
#include <hip/hip_runtime.h>
#include <hip/hip_bf16.h>

#define N_NODES 50000
#define N_EDGES 800000
#define IN_FEAT 1000
#define HID 128
#define HEADS 2
#define F1 (HEADS * HID) /* 256 */
#define NEG_SLOPE 0.2f
#define EPS_DENOM 1e-16f

typedef __attribute__((ext_vector_type(16))) _Float16 v16h;
typedef __attribute__((ext_vector_type(8))) float v8f;

// ---------------- helpers ----------------
__device__ __forceinline__ float leaky(float v) { return v > 0.f ? v : NEG_SLOPE * v; }

// monotone float<->uint encoding for atomicMax on floats
__device__ __forceinline__ unsigned fenc(float x) {
    unsigned b = __float_as_uint(x);
    return (b & 0x80000000u) ? ~b : (b | 0x80000000u);
}
__device__ __forceinline__ float fdec(unsigned u) {
    return __uint_as_float((u & 0x80000000u) ? (u & 0x7FFFFFFFu) : ~u);
}

// ---------------- Layer 1 GEMM: H1[M,256] = x[M,1000] @ W1[1000,256] ----------------
#define BM 128
#define BN 128
#define BK 32
#define KP 40 /* LDS row stride in halves: 80B rows, fragment offsets stay 16B aligned */

// Branch-free tile loaders: clamped addresses + cndmask zeroing (no EXEC divergence).
__device__ __forceinline__ void load_tiles(const float* __restrict__ x,
                                           const float* __restrict__ W1,
                                           int m0, int n0, int k0, int t,
                                           uint2 aReg[4], unsigned bReg[2][4]) {
    // A tile 128x32: float4 per load; x rows are 4000B -> every float4 is 16B aligned
#pragma unroll
    for (int i = 0; i < 4; i++) {
        int idx = t + i * 256;
        int row = idx >> 3;
        int c4 = (idx & 7) * 4;
        int gm = m0 + row;
        gm = gm < N_NODES ? gm : (N_NODES - 1); // row clamp (padding rows never stored)
        int gk = k0 + c4;
        bool ok = gk <= (IN_FEAT - 4); // whole float4 in-row (IN_FEAT % 4 == 0)
        int gkc = ok ? gk : (IN_FEAT - 4);
        float4 v = *(const float4*)(x + (size_t)gm * IN_FEAT + gkc);
        float f0 = ok ? v.x : 0.f, f1 = ok ? v.y : 0.f;
        float f2 = ok ? v.z : 0.f, f3 = ok ? v.w : 0.f;
        union { _Float16 f[2]; unsigned u; } p0, p1;
        p0.f[0] = (_Float16)f0; p0.f[1] = (_Float16)f1;
        p1.f[0] = (_Float16)f2; p1.f[1] = (_Float16)f3;
        aReg[i].x = p0.u; aReg[i].y = p1.u;
    }
    // B tile 32x128: two consecutive k-rows per iteration so the transposed LDS
    // store can pack (k,k+1) f16 pairs into b32 stores.
#pragma unroll
    for (int i = 0; i < 2; i++) {
        int idx = t + i * 256;
        int k2 = idx >> 5;          // 0..15 -> k pair {2k2, 2k2+1}
        int n4 = (idx & 31) * 4;
        int gk0 = k0 + 2 * k2, gk1 = gk0 + 1;
        bool ok0 = gk0 < IN_FEAT, ok1 = gk1 < IN_FEAT;
        int gc0 = ok0 ? gk0 : (IN_FEAT - 1);
        int gc1 = ok1 ? gk1 : (IN_FEAT - 1);
        float4 r0 = *(const float4*)(W1 + (size_t)gc0 * F1 + n0 + n4);
        float4 r1 = *(const float4*)(W1 + (size_t)gc1 * F1 + n0 + n4);
        float a0[4] = {r0.x, r0.y, r0.z, r0.w};
        float a1[4] = {r1.x, r1.y, r1.z, r1.w};
#pragma unroll
        for (int j = 0; j < 4; j++) {
            union { _Float16 f[2]; unsigned u; } p;
            p.f[0] = (_Float16)(ok0 ? a0[j] : 0.f);
            p.f[1] = (_Float16)(ok1 ? a1[j] : 0.f);
            bReg[i][j] = p.u;
        }
    }
}

__device__ __forceinline__ void store_tiles(_Float16* As, _Float16* Bs, int t,
                                            const uint2 aReg[4], const unsigned bReg[2][4]) {
#pragma unroll
    for (int i = 0; i < 4; i++) {
        int idx = t + i * 256;
        int row = idx >> 3, c4 = (idx & 7) * 4;
        *(uint2*)&As[row * KP + c4] = aReg[i]; // 8B-aligned ds_store_b64
    }
#pragma unroll
    for (int i = 0; i < 2; i++) {
        int idx = t + i * 256;
        int k2 = idx >> 5, n4 = (idx & 31) * 4;
#pragma unroll
        for (int j = 0; j < 4; j++)
            *(unsigned*)&Bs[(n4 + j) * KP + 2 * k2] = bReg[i][j]; // 4B-aligned ds_store_b32
    }
}

__global__ __launch_bounds__(256) void gemm1_wmma(const float* __restrict__ x,
                                                  const float* __restrict__ W1,
                                                  float* __restrict__ H1) {
    __shared__ __align__(16) _Float16 As[BM * KP];
    __shared__ __align__(16) _Float16 Bs[BN * KP];

    const int t = threadIdx.x;
    const int m0 = blockIdx.x * BM;
    const int n0 = blockIdx.y * BN;
    const int lane = t & 31;
    const int w = t >> 5;
    const int wrow = w >> 1;  // 0..3 -> 32-row slab
    const int wcol = w & 1;   // 0..1 -> 64-col slab

    v8f acc[2][4];
#pragma unroll
    for (int i = 0; i < 2; i++)
#pragma unroll
        for (int j = 0; j < 4; j++) acc[i][j] = (v8f){0.f, 0.f, 0.f, 0.f, 0.f, 0.f, 0.f, 0.f};

    const int NCHUNK = (IN_FEAT + BK - 1) / BK; // 32 (31 full + 8-wide tail, zero padded)
    uint2 aReg[4];
    unsigned bReg[2][4];
    load_tiles(x, W1, m0, n0, 0, t, aReg, bReg);

    for (int c = 0; c < NCHUNK; c++) {
        store_tiles(As, Bs, t, aReg, bReg);
        __syncthreads();
        if (c + 1 < NCHUNK) // prefetch next tile from global while WMMA consumes LDS
            load_tiles(x, W1, m0, n0, (c + 1) * BK, t, aReg, bReg);

        // Build fragments per ISA 16-bit layouts
        v16h af[2], bf[4];
#pragma unroll
        for (int i = 0; i < 2; i++) {
            int row = wrow * 32 + i * 16 + (lane & 15);
            int koff = (lane < 16) ? 0 : 8; // lanes 0-15: K{0-7,16-23}; 16-31: K{8-15,24-31}
            const _Float16* p = &As[row * KP + koff];
            ((uint4*)&af[i])[0] = *(const uint4*)p;
            ((uint4*)&af[i])[1] = *(const uint4*)(p + 16);
        }
#pragma unroll
        for (int j = 0; j < 4; j++) {
            int col = wcol * 64 + j * 16 + (lane & 15);
            int koff = (lane < 16) ? 0 : 8; // lanes 0-15: K{0-7,16-23}; 16-31: K{8-15,24-31}
            const _Float16* p = &Bs[col * KP + koff];
            ((uint4*)&bf[j])[0] = *(const uint4*)p;
            ((uint4*)&bf[j])[1] = *(const uint4*)(p + 16);
        }
#pragma unroll
        for (int i = 0; i < 2; i++)
#pragma unroll
            for (int j = 0; j < 4; j++)
                acc[i][j] = __builtin_amdgcn_wmma_f32_16x16x32_f16(
                    false, af[i], false, bf[j], (short)0, acc[i][j], false, false);
        __syncthreads();
    }

    // Store D: VGPR r -> rows r (lanes 0-15) / r+8 (lanes 16-31), N = lane&15
#pragma unroll
    for (int i = 0; i < 2; i++)
#pragma unroll
        for (int j = 0; j < 4; j++)
#pragma unroll
            for (int r = 0; r < 8; r++) {
                int gm = m0 + wrow * 32 + i * 16 + r + ((lane >= 16) ? 8 : 0);
                int gn = n0 + wcol * 64 + j * 16 + (lane & 15);
                if (gm < N_NODES) H1[(size_t)gm * F1 + gn] = acc[i][j][r];
            }
}

// ---------------- per-node attention projections, layer 1 ----------------
__global__ __launch_bounds__(256) void node_attn1(const float* __restrict__ H1,
                                                  const float* __restrict__ att_src,
                                                  const float* __restrict__ att_dst,
                                                  float* __restrict__ asrc,
                                                  float* __restrict__ adst) {
    __shared__ float ss[256], sd[256];
    const int n = blockIdx.x, t = threadIdx.x;
    float v = H1[(size_t)n * F1 + t];
    ss[t] = v * att_src[t]; // att flat [H*D], t = h*128+d
    sd[t] = v * att_dst[t];
    __syncthreads();
    for (int off = 64; off; off >>= 1) {
        if ((t & 127) < off) { ss[t] += ss[t + off]; sd[t] += sd[t + off]; }
        __syncthreads();
    }
    if (t == 0)        { asrc[n * 2 + 0] = ss[0];   adst[n * 2 + 0] = sd[0]; }
    else if (t == 128) { asrc[n * 2 + 1] = ss[128]; adst[n * 2 + 1] = sd[128]; }
}

// ---------------- edge softmax passes, layer 1 ----------------
__global__ void edge1_max(const int* __restrict__ ei, const float* __restrict__ asrc,
                          const float* __restrict__ adst, unsigned* __restrict__ m1) {
    int gid = blockIdx.x * blockDim.x + threadIdx.x;
    if (gid >= N_EDGES * HEADS) return;
    int e = gid >> 1, h = gid & 1;
    int s = ei[e], d = ei[N_EDGES + e];
    float v = leaky(asrc[s * 2 + h] + adst[d * 2 + h]);
    atomicMax(&m1[d * 2 + h], fenc(v));
}

__global__ void edge1_expsum(const int* __restrict__ ei, const float* __restrict__ asrc,
                             const float* __restrict__ adst, const unsigned* __restrict__ m1,
                             float* __restrict__ eexp, float* __restrict__ denom) {
    int gid = blockIdx.x * blockDim.x + threadIdx.x;
    if (gid >= N_EDGES * HEADS) return;
    int e = gid >> 1, h = gid & 1;
    int s = ei[e], d = ei[N_EDGES + e];
    float v = leaky(asrc[s * 2 + h] + adst[d * 2 + h]);
    float ex = __expf(v - fdec(m1[d * 2 + h]));
    eexp[e * 2 + h] = ex;
    atomicAdd(&denom[d * 2 + h], ex);
}

__global__ __launch_bounds__(256) void edge1_agg(const int* __restrict__ ei,
                                                 const float* __restrict__ H1,
                                                 const float* __restrict__ eexp,
                                                 const float* __restrict__ denom,
                                                 float* __restrict__ agg) {
    const int e = blockIdx.x, t = threadIdx.x, h = t >> 7;
    const int s = ei[e], d = ei[N_EDGES + e];
    float alpha = eexp[e * 2 + h] / (denom[d * 2 + h] + EPS_DENOM);
    float msg = H1[(size_t)s * F1 + t] * alpha;
    atomicAdd(&agg[(size_t)d * F1 + t], msg);
}

// ---------------- finalize layer 1 (+bias, relu) fused with layer-2 GEMV + attn ----------------
__global__ __launch_bounds__(256) void finalize_l1(const float* __restrict__ agg,
                                                   const float* __restrict__ b1,
                                                   const float* __restrict__ W2,
                                                   const float* __restrict__ att_src2,
                                                   const float* __restrict__ att_dst2,
                                                   float* __restrict__ h2,
                                                   float* __restrict__ as2,
                                                   float* __restrict__ ad2) {
    __shared__ float sr[256];
    const int n = blockIdx.x, t = threadIdx.x;
    float v = agg[(size_t)n * F1 + t] + b1[t];
    v = v > 0.f ? v : 0.f; // relu
    sr[t] = v * W2[t];     // W2: [256,1]
    __syncthreads();
    for (int off = 128; off; off >>= 1) {
        if (t < off) sr[t] += sr[t + off];
        __syncthreads();
    }
    if (t == 0) {
        float hv = sr[0];
        h2[n] = hv;
        as2[n] = hv * att_src2[0];
        ad2[n] = hv * att_dst2[0];
    }
}

// ---------------- edge softmax passes, layer 2 (scalar, 1 head) ----------------
__global__ void edge2_max(const int* __restrict__ ei, const float* __restrict__ as2,
                          const float* __restrict__ ad2, unsigned* __restrict__ m2) {
    int e = blockIdx.x * blockDim.x + threadIdx.x;
    if (e >= N_EDGES) return;
    int s = ei[e], d = ei[N_EDGES + e];
    atomicMax(&m2[d], fenc(leaky(as2[s] + ad2[d])));
}

__global__ void edge2_expsum(const int* __restrict__ ei, const float* __restrict__ as2,
                             const float* __restrict__ ad2, const unsigned* __restrict__ m2,
                             float* __restrict__ eexp2, float* __restrict__ denom2) {
    int e = blockIdx.x * blockDim.x + threadIdx.x;
    if (e >= N_EDGES) return;
    int s = ei[e], d = ei[N_EDGES + e];
    float ex = __expf(leaky(as2[s] + ad2[d]) - fdec(m2[d]));
    eexp2[e] = ex;
    atomicAdd(&denom2[d], ex);
}

__global__ void edge2_agg(const int* __restrict__ ei, const float* __restrict__ h2,
                          const float* __restrict__ eexp2, const float* __restrict__ denom2,
                          float* __restrict__ out2) {
    int e = blockIdx.x * blockDim.x + threadIdx.x;
    if (e >= N_EDGES) return;
    int s = ei[e], d = ei[N_EDGES + e];
    float alpha = eexp2[e] / (denom2[d] + EPS_DENOM);
    atomicAdd(&out2[d], h2[s] * alpha);
}

// ---------------- readout gather ----------------
__global__ void readout_k(const float* __restrict__ out2, const int* __restrict__ ridx,
                          const float* __restrict__ b2, float* __restrict__ out) {
    int i = blockIdx.x * blockDim.x + threadIdx.x;
    if (i < 1024) out[i] = out2[ridx[i]] + b2[0];
}

// ---------------- host launcher ----------------
extern "C" void kernel_launch(void* const* d_in, const int* in_sizes, int n_in,
                              void* d_out, int out_size, void* d_ws, size_t ws_size,
                              hipStream_t stream) {
    const float* x        = (const float*)d_in[0];
    const int*   ei       = (const int*)d_in[1];   // [2, E]
    const int*   ridx     = (const int*)d_in[2];   // [1024]
    const float* W1       = (const float*)d_in[3]; // [1000, 256]
    const float* att_src1 = (const float*)d_in[4]; // [2,128]
    const float* att_dst1 = (const float*)d_in[5];
    const float* b1       = (const float*)d_in[6]; // [256]
    const float* W2       = (const float*)d_in[7]; // [256,1]
    const float* att_src2 = (const float*)d_in[8]; // [1,1]
    const float* att_dst2 = (const float*)d_in[9];
    const float* b2       = (const float*)d_in[10];

    char* ws = (char*)d_ws;
    size_t off = 0;
    auto alloc = [&](size_t bytes) -> void* {
        void* p = ws + off;
        off += (bytes + 255) & ~(size_t)255;
        return p;
    };
    float*    H1     = (float*)alloc((size_t)N_NODES * F1 * 4);   // 51.2 MB
    float*    asrc1  = (float*)alloc((size_t)N_NODES * 2 * 4);
    float*    adst1  = (float*)alloc((size_t)N_NODES * 2 * 4);
    unsigned* m1     = (unsigned*)alloc((size_t)N_NODES * 2 * 4);
    float*    denom1 = (float*)alloc((size_t)N_NODES * 2 * 4);
    float*    eexp1  = (float*)alloc((size_t)N_EDGES * 2 * 4);    // 6.4 MB
    float*    agg1   = (float*)alloc((size_t)N_NODES * F1 * 4);   // 51.2 MB
    float*    h2     = (float*)alloc((size_t)N_NODES * 4);
    float*    as2    = (float*)alloc((size_t)N_NODES * 4);
    float*    ad2    = (float*)alloc((size_t)N_NODES * 4);
    unsigned* m2     = (unsigned*)alloc((size_t)N_NODES * 4);
    float*    denom2 = (float*)alloc((size_t)N_NODES * 4);
    float*    eexp2  = (float*)alloc((size_t)N_EDGES * 4);        // 3.2 MB
    float*    out2   = (float*)alloc((size_t)N_NODES * 4);

    // deterministic re-init of all accumulators (graph-capture safe)
    hipMemsetAsync(m1, 0, (size_t)N_NODES * 2 * 4, stream);     // enc(0)=most-negative key
    hipMemsetAsync(denom1, 0, (size_t)N_NODES * 2 * 4, stream);
    hipMemsetAsync(agg1, 0, (size_t)N_NODES * F1 * 4, stream);
    hipMemsetAsync(m2, 0, (size_t)N_NODES * 4, stream);
    hipMemsetAsync(denom2, 0, (size_t)N_NODES * 4, stream);
    hipMemsetAsync(out2, 0, (size_t)N_NODES * 4, stream);

    // Layer 1: WMMA GEMM
    dim3 g1((N_NODES + BM - 1) / BM, F1 / BN); // 391 x 2
    gemm1_wmma<<<g1, 256, 0, stream>>>(x, W1, H1);

    node_attn1<<<N_NODES, 256, 0, stream>>>(H1, att_src1, att_dst1, asrc1, adst1);

    const int ethreads = N_EDGES * HEADS;
    edge1_max<<<(ethreads + 255) / 256, 256, 0, stream>>>(ei, asrc1, adst1, m1);
    edge1_expsum<<<(ethreads + 255) / 256, 256, 0, stream>>>(ei, asrc1, adst1, m1, eexp1, denom1);
    edge1_agg<<<N_EDGES, 256, 0, stream>>>(ei, H1, eexp1, denom1, agg1);

    finalize_l1<<<N_NODES, 256, 0, stream>>>(agg1, b1, W2, att_src2, att_dst2, h2, as2, ad2);

    edge2_max<<<(N_EDGES + 255) / 256, 256, 0, stream>>>(ei, as2, ad2, m2);
    edge2_expsum<<<(N_EDGES + 255) / 256, 256, 0, stream>>>(ei, as2, ad2, m2, eexp2, denom2);
    edge2_agg<<<(N_EDGES + 255) / 256, 256, 0, stream>>>(ei, h2, eexp2, denom2, out2);

    readout_k<<<(1024 + 255) / 256, 256, 0, stream>>>(out2, ridx, b2, (float*)d_out);
}
